// FFNAttention_45569603011108
// MI455X (gfx1250) — matrix operand order repaired and checked by reading, not compiled, not run on hardware
//
#include <hip/hip_runtime.h>
#include <math.h>

// ---------------- problem constants ----------------
#define B_   16
#define S_   8192
#define D_   512
#define H_   512
#define SPLIT 64
#define CHUNK (S_ / SPLIT)       // 128 rows per score block
#define NEG_INF_ (-1e18f)

// ---------------- workspace layout (floats) ----------------
#define WS_V        0                         // v[512]  = Wk^T w_out
#define WS_BASE     512                       // base[16] = u.query[b] + w_out.(bq+bk)
#define PART_STRIDE 514                       // m, l, kacc[512]
#define WS_PART     528
#define WS_KAVG     (WS_PART + B_ * SPLIT * PART_STRIDE)  // kavg[16*512]

typedef __attribute__((ext_vector_type(16))) _Float16 v16h;
typedef __attribute__((ext_vector_type(8)))  float    v8f;

// ============================================================
// Kernel 1: v = Wk^T w_out ; base[b] = (Wq^T w_out).query[b] + w_out.(bq+bk)
// one block, 512 threads (16 waves)
// ============================================================
__global__ __launch_bounds__(512) void prep_kernel(
    const float* __restrict__ query,
    const float* __restrict__ Wq, const float* __restrict__ bq,
    const float* __restrict__ Wk, const float* __restrict__ bk,
    const float* __restrict__ w_out,
    float* __restrict__ ws) {
  __shared__ float u_sh[D_];
  __shared__ float sc_sh;
  const int t = threadIdx.x;                  // 0..511 == d index
  if (t == 0) sc_sh = 0.0f;
  __syncthreads();

  float vacc = 0.0f, uacc = 0.0f;
  for (int h = 0; h < H_; ++h) {
    const float w = w_out[h];
    vacc += w * Wk[h * D_ + t];
    uacc += w * Wq[h * D_ + t];
  }
  ws[WS_V + t] = vacc;
  u_sh[t] = uacc;
  atomicAdd(&sc_sh, w_out[t] * (bq[t] + bk[t]));
  __syncthreads();

  const int wave = t >> 5;                    // 16 waves == 16 batches
  const int lane = t & 31;
  float p = 0.0f;
  for (int d = lane; d < D_; d += 32) p += u_sh[d] * query[wave * D_ + d];
  for (int off = 16; off; off >>= 1) p += __shfl_xor(p, off, 32);
  if (lane == 0) ws[WS_BASE + wave] = p + sc_sh;
}

// ============================================================
// Kernel 2: streaming score + chunk-local softmax partials
// grid = B_*SPLIT blocks, 256 threads (8 waves, 16 rows/wave)
// ============================================================
__global__ __launch_bounds__(256) void score_kernel(
    const float* __restrict__ key, const int* __restrict__ mask,
    float* ws, float* attnw_out /* = d_out + B_*H_ */) {
  __shared__ float sc[CHUNK];
  __shared__ float kacc_sh[D_];
  __shared__ float lsum_sh;

  const int bid  = blockIdx.x;
  const int b    = bid / SPLIT;
  const int s0   = (bid % SPLIT) * CHUNK;
  const int t    = threadIdx.x;
  const int wave = t >> 5;
  const int lane = t & 31;

  kacc_sh[t]       = 0.0f;
  kacc_sh[t + 256] = 0.0f;
  if (t == 0) lsum_sh = 0.0f;

  // per-lane slice of v (16 contiguous d's)
  const float4* vp = (const float4*)(ws + WS_V + lane * 16);
  const float4 v0 = vp[0], v1 = vp[1], v2 = vp[2], v3 = vp[3];
  const float base_b = ws[WS_BASE + b];

  // ---- pass 1: scores ----
  for (int rr = 0; rr < 16; ++rr) {
    const int r = wave * 16 + rr;
    const int s = s0 + r;
    const float4* kp =
        (const float4*)(key + ((size_t)b * S_ + s) * D_ + lane * 16);
    if (rr + 1 < 16)
      __builtin_prefetch((const char*)kp + D_ * sizeof(float), 0, 0);
    const float4 k0 = kp[0], k1 = kp[1], k2 = kp[2], k3 = kp[3];
    float d = k0.x * v0.x + k0.y * v0.y + k0.z * v0.z + k0.w * v0.w
            + k1.x * v1.x + k1.y * v1.y + k1.z * v1.z + k1.w * v1.w
            + k2.x * v2.x + k2.y * v2.y + k2.z * v2.z + k2.w * v2.w
            + k3.x * v3.x + k3.y * v3.y + k3.z * v3.z + k3.w * v3.w;
    for (int off = 16; off; off >>= 1) d += __shfl_xor(d, off, 32);
    if (lane == 0) {
      float w = d + base_b;
      if (mask[(size_t)b * S_ + s] != 0) w = NEG_INF_;
      sc[r] = w;
      attnw_out[(size_t)b * S_ + s] = w;
    }
  }
  __syncthreads();

  // ---- chunk max (redundant per-thread; LDS reads are cheap) ----
  float m = -3.0e38f;
  for (int i = 0; i < CHUNK; ++i) m = fmaxf(m, sc[i]);
  __syncthreads();

  // ---- p = exp(w - m), chunk sum ----
  if (t < CHUNK) {
    const float p = __expf(sc[t] - m);
    sc[t] = p;
    atomicAdd(&lsum_sh, p);
  }
  __syncthreads();

  // ---- pass 2: kacc = sum_s p_s * key[s,:]  (chunk is L2-hot) ----
  float4 a0 = {0, 0, 0, 0}, a1 = {0, 0, 0, 0}, a2 = {0, 0, 0, 0},
         a3 = {0, 0, 0, 0};
  for (int rr = 0; rr < 16; ++rr) {
    const int r = wave * 16 + rr;
    const int s = s0 + r;
    const float p = sc[r];
    const float4* kp =
        (const float4*)(key + ((size_t)b * S_ + s) * D_ + lane * 16);
    const float4 k0 = kp[0], k1 = kp[1], k2 = kp[2], k3 = kp[3];
    a0.x += p * k0.x; a0.y += p * k0.y; a0.z += p * k0.z; a0.w += p * k0.w;
    a1.x += p * k1.x; a1.y += p * k1.y; a1.z += p * k1.z; a1.w += p * k1.w;
    a2.x += p * k2.x; a2.y += p * k2.y; a2.z += p * k2.z; a2.w += p * k2.w;
    a3.x += p * k3.x; a3.y += p * k3.y; a3.z += p * k3.z; a3.w += p * k3.w;
  }
  const int dbase = lane * 16;
  atomicAdd(&kacc_sh[dbase +  0], a0.x); atomicAdd(&kacc_sh[dbase +  1], a0.y);
  atomicAdd(&kacc_sh[dbase +  2], a0.z); atomicAdd(&kacc_sh[dbase +  3], a0.w);
  atomicAdd(&kacc_sh[dbase +  4], a1.x); atomicAdd(&kacc_sh[dbase +  5], a1.y);
  atomicAdd(&kacc_sh[dbase +  6], a1.z); atomicAdd(&kacc_sh[dbase +  7], a1.w);
  atomicAdd(&kacc_sh[dbase +  8], a2.x); atomicAdd(&kacc_sh[dbase +  9], a2.y);
  atomicAdd(&kacc_sh[dbase + 10], a2.z); atomicAdd(&kacc_sh[dbase + 11], a2.w);
  atomicAdd(&kacc_sh[dbase + 12], a3.x); atomicAdd(&kacc_sh[dbase + 13], a3.y);
  atomicAdd(&kacc_sh[dbase + 14], a3.z); atomicAdd(&kacc_sh[dbase + 15], a3.w);
  __syncthreads();

  // ---- write partial ----
  float* part = ws + WS_PART + (size_t)bid * PART_STRIDE;
  if (t == 0) { part[0] = m; part[1] = lsum_sh; }
  part[2 + t]       = kacc_sh[t];
  part[2 + t + 256] = kacc_sh[t + 256];
}

// ============================================================
// Kernel 3: merge SPLIT softmax partials -> kavg[b, :]
// grid = B_ blocks, 256 threads
// ============================================================
__global__ __launch_bounds__(256) void combine_kernel(
    const float* __restrict__ ws, float* __restrict__ kavg) {
  __shared__ float e_sh[SPLIT];
  __shared__ float L_sh;
  const int b = blockIdx.x, t = threadIdx.x;
  if (t == 0) L_sh = 0.0f;

  float M = -3.0e38f;
  for (int i = 0; i < SPLIT; ++i)
    M = fmaxf(M, ws[WS_PART + (size_t)(b * SPLIT + i) * PART_STRIDE]);
  __syncthreads();

  if (t < SPLIT) {
    const float* p = ws + WS_PART + (size_t)(b * SPLIT + t) * PART_STRIDE;
    const float e = __expf(p[0] - M);
    e_sh[t] = e;
    atomicAdd(&L_sh, p[1] * e);
  }
  __syncthreads();

  const float inv = 1.0f / L_sh;
  for (int d = t; d < D_; d += 256) {
    float a = 0.0f;
    for (int i = 0; i < SPLIT; ++i)
      a += ws[WS_PART + (size_t)(b * SPLIT + i) * PART_STRIDE + 2 + d] * e_sh[i];
    kavg[b * D_ + d] = a * inv;
  }
}

// ============================================================
// Kernel 4: attn[16 x 512] = kavg[16 x 512] @ Wk^T + bk   (WMMA)
// Paired-f16 split (hi+lo, 3 WMMAs) for near-fp32 accuracy.
// grid = H_/16 blocks, 32 threads (one wave / 16-col tile, EXEC all-1s)
// ============================================================
__global__ __launch_bounds__(32) void proj_kernel(
    const float* __restrict__ kavg, const float* __restrict__ Wk,
    const float* __restrict__ bk, float* __restrict__ attn_out) {
  const int lane  = threadIdx.x;
  const int n0    = blockIdx.x * 16;
  const int row16 = lane & 15;     // A row (=batch) for A; B/C/D column for B
  const int hi    = lane >> 4;     // lane half select

  v8f acc = {};
  for (int k0 = 0; k0 < D_; k0 += 32) {
    // A (16x32 f16 layout): elements e -> K = k0 + hi*16 + e
    //                       (matches B ordering)
    const float* arow = kavg + row16 * D_;
    const int ka0 = k0 + hi * 16;
    const int ka1 = ka0 + 8;
    // B (32x16 f16 layout): column = lane&15, elements e -> K = k0 + hi*16 + e
    const float* brow = Wk + (n0 + row16) * D_;
    const int kb0 = k0 + hi * 16;

    v16h ahi, alo, bhi, blo;
#pragma unroll
    for (int e = 0; e < 8; ++e) {
      const float a0f = arow[ka0 + e];
      const _Float16 h0 = (_Float16)a0f;
      ahi[e] = h0; alo[e] = (_Float16)(a0f - (float)h0);
      const float a1f = arow[ka1 + e];
      const _Float16 h1 = (_Float16)a1f;
      ahi[e + 8] = h1; alo[e + 8] = (_Float16)(a1f - (float)h1);
    }
#pragma unroll
    for (int e = 0; e < 16; ++e) {
      const float bf = brow[kb0 + e];
      const _Float16 h = (_Float16)bf;
      bhi[e] = h; blo[e] = (_Float16)(bf - (float)h);
    }
    // (Ahi+Alo)(Bhi+Blo) ~= Ahi*Bhi + Ahi*Blo + Alo*Bhi
    acc = __builtin_amdgcn_wmma_f32_16x16x32_f16(false, ahi, false, bhi,
                                                 (short)0, acc, false, false);
    acc = __builtin_amdgcn_wmma_f32_16x16x32_f16(false, ahi, false, blo,
                                                 (short)0, acc, false, false);
    acc = __builtin_amdgcn_wmma_f32_16x16x32_f16(false, alo, false, bhi,
                                                 (short)0, acc, false, false);
  }

  const float bias = bk[n0 + row16];
#pragma unroll
  for (int r = 0; r < 8; ++r) {
    const int m = hi * 8 + r;                 // C/D layout: lanes>=16 -> M=8+r
    attn_out[m * H_ + n0 + row16] = acc[r] + bias;
  }
}

// ============================================================
extern "C" void kernel_launch(void* const* d_in, const int* in_sizes, int n_in,
                              void* d_out, int out_size, void* d_ws,
                              size_t ws_size, hipStream_t stream) {
  const float* query = (const float*)d_in[0];
  const float* key   = (const float*)d_in[1];
  const int*   mask  = (const int*)d_in[2];
  const float* Wq    = (const float*)d_in[3];
  const float* bq    = (const float*)d_in[4];
  const float* Wk    = (const float*)d_in[5];
  const float* bk    = (const float*)d_in[6];
  const float* w_out = (const float*)d_in[7];
  float* out = (float*)d_out;          // [attn: B*H][attn_weight: B*S]
  float* ws  = (float*)d_ws;           // ~2.1 MB of floats used

  prep_kernel<<<1, 512, 0, stream>>>(query, Wq, bq, Wk, bk, w_out, ws);
  score_kernel<<<B_ * SPLIT, 256, 0, stream>>>(key, mask, ws, out + B_ * H_);
  combine_kernel<<<B_, 256, 0, stream>>>(ws, ws + WS_KAVG);
  proj_kernel<<<H_ / 16, 32, 0, stream>>>(ws + WS_KAVG, Wk, bk, out);
}
